// LSTM3_65781719106001
// MI455X (gfx1250) — matrix operand order repaired and checked by reading, not compiled, not run on hardware
//
#include <hip/hip_runtime.h>
#include <hip/hip_bf16.h>

// MI455X / gfx1250 persistent fused 2-layer LSTM + MLP head.
// One workgroup per 64-row batch tile; weights live in VGPRs as WMMA
// B-fragments, state lives in LDS, x streamed with a 1-step register
// pipeline + global_prefetch.

typedef __attribute__((ext_vector_type(16))) _Float16 v16h;
typedef __attribute__((ext_vector_type(8)))  float    v8f;

#define TSTEPS 512
#define BATCH  512
#define FEAT   14
#define H1     32
#define H2     64
#define BT     64          // batch rows per workgroup
#define NWG    (BATCH / BT)
#define NTHR   256         // 8 wave32 waves

__device__ __forceinline__ float fsig(float x)  { return 1.0f / (1.0f + __expf(-x)); }
__device__ __forceinline__ float ftanh(float x) { return 2.0f * fsig(2.0f * x) - 1.0f; }
__device__ __forceinline__ float frelu(float x) { return x > 0.0f ? x : 0.0f; }

// ---- WMMA fragment helpers (A/B in contiguous K order k=16h+s, C/D per ISA, wave32) ----------------
// A: 16x32 f16 (MxK). lane<16: row=lane, K=0..15; lane>=16:
// row=lane-16, K=16..31. VGPR pairs pack consecutive K.
__device__ __forceinline__ v16h load_A(const _Float16* S, int ld, int mrow,
                                       int kbase, int lane) {
  const _Float16* p = S + (mrow + (lane & 15)) * ld + kbase;
  const int kh = (lane >> 4) << 4;   // +16 for hi half-wave
  v16h a;
#pragma unroll
  for (int v = 0; v < 4; ++v) {
    a[2 * v]     = p[kh + 2 * v];
    a[2 * v + 1] = p[kh + 2 * v + 1];
  }
#pragma unroll
  for (int v = 0; v < 4; ++v) {
    a[8 + 2 * v] = p[8 + kh + 2 * v];
    a[9 + 2 * v] = p[8 + kh + 2 * v + 1];
  }
  return a;
}

// B: 32x16 f16 (KxN), stored row-major [K][ldn] in LDS. lane<16: col=lane,
// K=0..15; lane>=16: col=lane-16, K=16..31. Element e of v16h <-> K.
__device__ __forceinline__ v16h load_B(const _Float16* W, int ldn, int ncol,
                                       int kbase, int lane) {
  const _Float16* p = W + (size_t)(kbase + ((lane >> 4) << 4)) * ldn
                        + ncol + (lane & 15);
  v16h b;
#pragma unroll
  for (int e = 0; e < 16; ++e) b[e] = p[e * ldn];
  return b;
}

// C/D: 16x16 f32. VGPR v: lanes 0-15 -> M=v, lanes 16-31 -> M=v+8; N=lane&15.
__device__ __forceinline__ void store_C(float* G, int ldn, int mrow, int ncol,
                                        int lane, v8f c) {
  const int n  = ncol + (lane & 15);
  const int m0 = mrow + ((lane >> 4) << 3);
#pragma unroll
  for (int v = 0; v < 8; ++v) G[(m0 + v) * ldn + n] = c[v];
}

__device__ __forceinline__ v8f bcast8(float v) {
  v8f r;
#pragma unroll
  for (int i = 0; i < 8; ++i) r[i] = v;
  return r;
}

#define WMMA(A_, B_, C_)                                                     \
  __builtin_amdgcn_wmma_f32_16x16x32_f16(false, (A_), false, (B_),           \
                                         (short)0, (C_), false, false)

// LDS layout (124,416 B total; <= 320KB/WGP so 2 WGs/WGP still fit)
#define SM_G1    0        // 64x128 f32  (layer-1 gate preacts)      32768
#define SM_G2    32768    // 64x256 f32  (layer-2 preacts + staging) 65536
#define SM_C1    98304    // 64x32  f32                               8192
#define SM_H1    106496   // 64x32  f16                               4096
#define SM_H2    110592   // 64x64  f16                               8192
#define SM_XS    118784   // 64x32  f16 (x tile, K-padded 14->32)     4096
#define SM_B1    122880   // 128 f32                                   512
#define SM_B2    123392   // 256 f32                                  1024
#define SMEM_BYTES 124416

__global__ void __launch_bounds__(NTHR)
lstm3_persistent(const float* __restrict__ x,
                 const float* __restrict__ W_ih1, const float* __restrict__ W_hh1,
                 const float* __restrict__ b_ih1, const float* __restrict__ b_hh1,
                 const float* __restrict__ W_ih2, const float* __restrict__ W_hh2,
                 const float* __restrict__ b_ih2, const float* __restrict__ b_hh2,
                 const float* __restrict__ W_fc1, const float* __restrict__ b_fc1,
                 const float* __restrict__ W_fc2, const float* __restrict__ b_fc2,
                 const float* __restrict__ W_fc,  const float* __restrict__ b_fc,
                 float* __restrict__ out) {
  extern __shared__ char smem[];
  float*    g1    = (float*)(smem + SM_G1);
  float*    g2    = (float*)(smem + SM_G2);
  float*    c1s   = (float*)(smem + SM_C1);
  _Float16* h1s   = (_Float16*)(smem + SM_H1);
  _Float16* h2s   = (_Float16*)(smem + SM_H2);
  _Float16* xs    = (_Float16*)(smem + SM_XS);
  float*    bias1 = (float*)(smem + SM_B1);
  float*    bias2 = (float*)(smem + SM_B2);

  const int tid   = threadIdx.x;
  const int lane  = tid & 31;
  const int wid   = tid >> 5;           // wave id 0..7
  const int bbase = blockIdx.x * BT;

  // ---------------- init state, biases ----------------
  for (int i = tid; i < BT * H1; i += NTHR) {
    c1s[i] = 0.0f; h1s[i] = (_Float16)0.0f; xs[i] = (_Float16)0.0f;
  }
  for (int i = tid; i < BT * H2; i += NTHR) h2s[i] = (_Float16)0.0f;
  if (tid < 128) bias1[tid] = b_ih1[tid] + b_hh1[tid];
  bias2[tid] = b_ih2[tid] + b_hh2[tid];           // 256 threads exactly

  // ---------------- stage weights -> per-wave register B-fragments -------
  _Float16* wstage = (_Float16*)g2;               // scratch (<= 32KB f16)

  // W_ih1^T, K padded 14->32: [32][128]
  for (int i = tid; i < 32 * 128; i += NTHR) {
    int k = i >> 7, n = i & 127;
    wstage[i] = (k < FEAT) ? (_Float16)W_ih1[n * FEAT + k] : (_Float16)0.0f;
  }
  __syncthreads();
  v16h Bx = load_B(wstage, 128, 16 * wid, 0, lane);
  __syncthreads();

  // W_hh1^T: [32][128]
  for (int i = tid; i < 32 * 128; i += NTHR) {
    int k = i >> 7, n = i & 127;
    wstage[i] = (_Float16)W_hh1[n * H1 + k];
  }
  __syncthreads();
  v16h B1 = load_B(wstage, 128, 16 * wid, 0, lane);
  __syncthreads();

  // W_ih2^T: [32][256]
  for (int i = tid; i < 32 * 256; i += NTHR) {
    int k = i >> 8, n = i & 255;
    wstage[i] = (_Float16)W_ih2[n * H1 + k];
  }
  __syncthreads();
  v16h B2a0 = load_B(wstage, 256, 16 * (2 * wid + 0), 0, lane);
  v16h B2a1 = load_B(wstage, 256, 16 * (2 * wid + 1), 0, lane);
  __syncthreads();

  // W_hh2^T: [64][256]
  for (int i = tid; i < 64 * 256; i += NTHR) {
    int k = i >> 8, n = i & 255;
    wstage[i] = (_Float16)W_hh2[n * H2 + k];
  }
  __syncthreads();
  v16h B2b00 = load_B(wstage, 256, 16 * (2 * wid + 0), 0,  lane);
  v16h B2b01 = load_B(wstage, 256, 16 * (2 * wid + 0), 32, lane);
  v16h B2b10 = load_B(wstage, 256, 16 * (2 * wid + 1), 0,  lane);
  v16h B2b11 = load_B(wstage, 256, 16 * (2 * wid + 1), 32, lane);
  __syncthreads();

  // ---------------- time loop ----------------
  float xr[4];                                    // x(t) register pipeline
  {
    const float* xp = x + (size_t)bbase * FEAT;   // t = 0
#pragma unroll
    for (int u = 0; u < 4; ++u) {
      int e = u * NTHR + tid;
      xr[u] = (e < BT * FEAT) ? xp[e] : 0.0f;
    }
  }

  for (int t = 0; t < TSTEPS; ++t) {
    // stage x(t) regs -> LDS (cols 14..31 stay zero from init)
#pragma unroll
    for (int u = 0; u < 4; ++u) {
      int e = u * NTHR + tid;
      if (e < BT * FEAT) {
        int r = e / FEAT, c = e - r * FEAT;
        xs[r * 32 + c] = (_Float16)xr[u];
      }
    }
    __syncthreads();                              // (0) xs ready, h2s(t-1) ready

    // preload x(t+1); prefetch x(t+2) tile into cache
    if (t + 1 < TSTEPS) {
      const float* xp = x + ((size_t)(t + 1) * BATCH + bbase) * FEAT;
#pragma unroll
      for (int u = 0; u < 4; ++u) {
        int e = u * NTHR + tid;
        xr[u] = (e < BT * FEAT) ? xp[e] : 0.0f;
      }
    }
    if (t + 2 < TSTEPS && tid < 28) {
      const char* pp = (const char*)(x + ((size_t)(t + 2) * BATCH + bbase) * FEAT);
      __builtin_prefetch(pp + tid * 128, 0, 1);   // global_prefetch_b8
    }

    // ---- layer 1: g1 = bias1 + x*W_ih1^T + h1*W_hh1^T  (wave w -> N-tile w)
    {
      const int   nc = 16 * wid;
      const float bv = bias1[nc + (lane & 15)];
#pragma unroll
      for (int mt = 0; mt < 4; ++mt) {
        v8f  acc = bcast8(bv);
        v16h ax  = load_A(xs,  32, mt * 16, 0, lane);
        acc = WMMA(ax, Bx, acc);
        v16h ah  = load_A(h1s, 32, mt * 16, 0, lane);
        acc = WMMA(ah, B1, acc);
        store_C(g1, 128, mt * 16, nc, lane, acc);
      }
    }
    __syncthreads();                              // (1) g1 ready

    // ---- layer-1 pointwise: PyTorch gate order i,f,g,o
#pragma unroll
    for (int u = 0; u < 8; ++u) {
      int idx = u * NTHR + tid;                   // = b*32 + j
      int b = idx >> 5, j = idx & 31;
      float gi = g1[b * 128 + j];
      float gf = g1[b * 128 + 32 + j];
      float gg = g1[b * 128 + 64 + j];
      float go = g1[b * 128 + 96 + j];
      float c  = fsig(gf) * c1s[idx] + fsig(gi) * ftanh(gg);
      c1s[idx] = c;
      h1s[idx] = (_Float16)(fsig(go) * ftanh(c));
    }
    __syncthreads();                              // (2) h1 ready

    // ---- layer 2: g2 = bias2 + h1*W_ih2^T + h2*W_hh2^T (wave w -> tiles 2w,2w+1)
    {
#pragma unroll
      for (int mt = 0; mt < 4; ++mt) {
        v16h a1  = load_A(h1s, 32, mt * 16, 0,  lane);
        v16h a20 = load_A(h2s, 64, mt * 16, 0,  lane);
        v16h a21 = load_A(h2s, 64, mt * 16, 32, lane);
#pragma unroll
        for (int j = 0; j < 2; ++j) {
          const int   nc = 16 * (2 * wid + j);
          const float bv = bias2[nc + (lane & 15)];
          v8f acc = bcast8(bv);
          acc = WMMA(a1,  j ? B2a1  : B2a0,  acc);
          acc = WMMA(a20, j ? B2b10 : B2b00, acc);
          acc = WMMA(a21, j ? B2b11 : B2b01, acc);
          store_C(g2, 256, mt * 16, nc, lane, acc);
        }
      }
    }
    __syncthreads();                              // (3) g2 ready

    // ---- layer-2 pointwise (incoming c2 == 0 in reference -> f-gate dead)
#pragma unroll
    for (int u = 0; u < 16; ++u) {
      int idx = u * NTHR + tid;                   // = b*64 + j
      int b = idx >> 6, j = idx & 63;
      float gi = g2[b * 256 + j];
      float gg = g2[b * 256 + 128 + j];
      float go = g2[b * 256 + 192 + j];
      float c2 = fsig(gi) * ftanh(gg);
      h2s[idx] = (_Float16)(fsig(go) * ftanh(c2));
    }
    // barrier (0) of next iteration orders h2s before its next readers
  }
  __syncthreads();

  // ---------------- MLP head on final h2 ----------------
  if (tid < BT) {
    float v[H2];
#pragma unroll
    for (int k = 0; k < H2; ++k) v[k] = frelu((float)h2s[tid * H2 + k]);
    float a1[8];
#pragma unroll
    for (int i = 0; i < 8; ++i) {
      float s = b_fc1[i];
#pragma unroll
      for (int k = 0; k < H2; ++k) s += W_fc1[i * H2 + k] * v[k];
      a1[i] = frelu(s);
    }
    float a2[8];
#pragma unroll
    for (int i = 0; i < 8; ++i) {
      float s = b_fc2[i];
#pragma unroll
      for (int k = 0; k < 8; ++k) s += W_fc2[i * 8 + k] * a1[k];
      a2[i] = frelu(s);
    }
    float o = b_fc[0];
#pragma unroll
    for (int k = 0; k < 8; ++k) o += W_fc[k] * a2[k];
    out[bbase + tid] = o;
  }
}

extern "C" void kernel_launch(void* const* d_in, const int* in_sizes, int n_in,
                              void* d_out, int out_size, void* d_ws, size_t ws_size,
                              hipStream_t stream) {
  (void)in_sizes; (void)n_in; (void)d_ws; (void)ws_size; (void)out_size;
  const float* x     = (const float*)d_in[0];
  const float* W_ih1 = (const float*)d_in[1];
  const float* W_hh1 = (const float*)d_in[2];
  const float* b_ih1 = (const float*)d_in[3];
  const float* b_hh1 = (const float*)d_in[4];
  const float* W_ih2 = (const float*)d_in[5];
  const float* W_hh2 = (const float*)d_in[6];
  const float* b_ih2 = (const float*)d_in[7];
  const float* b_hh2 = (const float*)d_in[8];
  const float* W_fc1 = (const float*)d_in[9];
  const float* b_fc1 = (const float*)d_in[10];
  const float* W_fc2 = (const float*)d_in[11];
  const float* b_fc2 = (const float*)d_in[12];
  const float* W_fc  = (const float*)d_in[13];
  const float* b_fc  = (const float*)d_in[14];

  lstm3_persistent<<<dim3(NWG), dim3(NTHR), SMEM_BYTES, stream>>>(
      x, W_ih1, W_hh1, b_ih1, b_hh1, W_ih2, W_hh2, b_ih2, b_hh2,
      W_fc1, b_fc1, W_fc2, b_fc2, W_fc, b_fc, (float*)d_out);
}
